// NN_each_LN_exp_10359461118396
// MI455X (gfx1250) — matrix operand order repaired and checked by reading, not compiled, not run on hardware
//
#include <hip/hip_runtime.h>
#include <hip/hip_bf16.h>

typedef __attribute__((ext_vector_type(16))) _Float16 v16h;
typedef __attribute__((ext_vector_type(8)))  float    v8f;

#define NBOARD 65536
#define WAVES 8
#define TPB 256

// ---- pre-swizzled B-operand blob layout (halfword units) ----
// conv mats: K=96 (3 chunks), N=80 (5 tiles)  -> 3*5*512 = 7680 hw each
// W1: K=96 (3), N=112 (7) -> 10752 ; W2: K=128 (4), N=112 (7) -> 14336 ; W3: K=128 (4), N=16 (1) -> 2048
#define OFF_EACH   0
#define OFF_NOT    7680
#define OFF_NOT2  15360
#define OFF_EMPTY 23040
#define OFF_W1    30720
#define OFF_W2    41472
#define OFF_W3    55808
#define BLOB_HW   57856
#define BLOB_BYTES (BLOB_HW * 2)      // 115712
#define BIAS_FLOATS 240               // b1[112] b2[112] b3+pad[16]
#define WS_BYTES (BLOB_BYTES + BIAS_FLOATS * 4)   // 116672 (16B multiple)

// ---- LDS layout ----
#define KSTR 80     // kind row stride (bytes)
#define FSTR 84     // E / acv2 row stride (floats)  (odd dword multiple -> no lane-group bank alias)
#define HSTR 116    // hidden row stride (floats)
#define PW_KIND 0
#define PW_E    1280                    // 16*KSTR
#define PW_ACV  (PW_E   + 16*FSTR*4)    // 6656
#define PW_H    (PW_ACV + 16*FSTR*4)    // 12032
#define PW_SIZE (PW_H   + 16*HSTR*4)    // 19456
#define LDS_BIAS  BLOB_BYTES
#define LDS_PW0   (LDS_BIAS + BIAS_FLOATS*4)       // 116672
#define LDS_TOTAL (LDS_PW0 + WAVES*PW_SIZE)        // 272320 (< 320KB WGP LDS)

// A-matrix (16x32 f16) per-lane K index: VGPR v half h, lanes0-15: K=2v+h(+16 for v>=4), lanes16-31: +8
__device__ __forceinline__ int kidxA(int t, int g) {
  int v = t >> 1, h = t & 1;
  return (v < 4) ? (8*g + 2*v + h) : (16 + 8*g + 2*(v-4) + h);
}

__device__ __forceinline__ v8f wmma16(v16h a, v16h b, v8f c) {
  return __builtin_amdgcn_wmma_f32_16x16x32_f16(false, a, false, b, (short)0, c, false, false);
}

// Build swizzled f16 B-operands (implied 72x72 conv matrices + MLP weights) and padded biases into d_ws.
__global__ void prep_kernel(const float* __restrict__ w_each, const float* __restrict__ w_not,
                            const float* __restrict__ w_not2, const float* __restrict__ w_empty,
                            const float* __restrict__ W1, const float* __restrict__ W2,
                            const float* __restrict__ W3,
                            const float* __restrict__ b1, const float* __restrict__ b2,
                            const float* __restrict__ b3, void* __restrict__ ws) {
  int tid = blockIdx.x * blockDim.x + threadIdx.x;
  _Float16* blob = (_Float16*)ws;
  if (tid < BLOB_HW) {
    int rem = tid;
    const float* w; int NT, type, Nmax, Kmax;
    if      (rem < OFF_NOT)   { w = w_each;  rem -= OFF_EACH;  NT = 5; type = 0; Nmax = 72;  Kmax = 72;  }
    else if (rem < OFF_NOT2)  { w = w_not;   rem -= OFF_NOT;   NT = 5; type = 0; Nmax = 72;  Kmax = 72;  }
    else if (rem < OFF_EMPTY) { w = w_not2;  rem -= OFF_NOT2;  NT = 5; type = 0; Nmax = 72;  Kmax = 72;  }
    else if (rem < OFF_W1)    { w = w_empty; rem -= OFF_EMPTY; NT = 5; type = 0; Nmax = 72;  Kmax = 72;  }
    else if (rem < OFF_W2)    { w = W1;      rem -= OFF_W1;    NT = 7; type = 1; Nmax = 100; Kmax = 72;  }
    else if (rem < OFF_W3)    { w = W2;      rem -= OFF_W2;    NT = 7; type = 1; Nmax = 100; Kmax = 100; }
    else                      { w = W3;      rem -= OFF_W3;    NT = 1; type = 1; Nmax = 1;   Kmax = 100; }
    int t  = rem & 15;
    int l  = (rem >> 4) & 31;
    int bi = rem >> 9;
    int kc = bi / NT, nt = bi % NT;
    int g  = l >> 4,  nl = l & 15;
    int k  = kc*32 + ((t < 8) ? (8*g + t) : (16 + 8*g + (t - 8)));   // B-matrix (32x16) lane layout: lanes0-15 K=0..7,16..23, lanes16-31 K=8..15,24..31
    int n  = nt*16 + nl;
    float val = 0.0f;
    if (k < Kmax && n < Nmax) {
      if (type == 0) {
        // B[k][n] = Mconv[n][k]; out cell n=(i,j), in cell k=(p,q)
        int p = k / 6, q = k % 6, i = n / 6, j = n % 6;
        int dy = 6 + p - i, dx = 6 + q - j;
        if (dy >= 0 && dy < 13 && dx >= 0 && dx < 13) val = w[dy*13 + dx];
      } else {
        val = w[n*Kmax + k];     // B[k][n] = W[n][k]
      }
    }
    blob[tid] = (_Float16)val;
  }
  int bt = tid - BLOB_HW;
  if (bt >= 0 && bt < BIAS_FLOATS) {
    float v = 0.0f;
    if (bt < 112)      { if (bt < 100) v = b1[bt]; }
    else if (bt < 224) { int i = bt - 112; if (i < 100) v = b2[i]; }
    else               { if (bt == 224) v = b3[0]; }
    ((float*)((char*)ws + BLOB_BYTES))[bt] = v;
  }
}

__global__ __launch_bounds__(TPB, 1)
void board_mlp_kernel(const int* __restrict__ dots,
                      const float* __restrict__ b_each_p, const float* __restrict__ b_not_p,
                      const float* __restrict__ b_not2_p, const float* __restrict__ b_empty_p,
                      const void* __restrict__ ws, float* __restrict__ out) {
  extern __shared__ char smem[];
  const int tid  = threadIdx.x;
  const int lane = tid & 31;
  const int wave = tid >> 5;
  const int g    = lane >> 4;
  const int nl   = lane & 15;

  // stage weights + biases into LDS (fully coalesced 16B copies)
  {
    const float4* s4 = (const float4*)ws;
    float4* d4 = (float4*)smem;
    for (int i = tid; i < WS_BYTES/16; i += TPB) d4[i] = s4[i];
  }
  __syncthreads();

  _Float16* blob  = (_Float16*)smem;
  const float* biasL = (const float*)(smem + LDS_BIAS);
  char* pw = smem + LDS_PW0 + wave * PW_SIZE;
  unsigned char* kindL = (unsigned char*)(pw + PW_KIND);  // [16][KSTR] u8
  float* EL = (float*)(pw + PW_E);                        // empty-conv   [16][FSTR]
  float* AL = (float*)(pw + PW_ACV);                      // acv2 carry   [16][FSTR]
  float* HL = (float*)(pw + PW_H);                        // MLP hidden   [16][HSTR]

  const int boardBase = blockIdx.x * (WAVES*16) + wave*16;

  // load 72 trimmed cells x 16 boards: lanes split (cell parity = lane group, board = lane%16)
  #pragma unroll 6
  for (int it = 0; it < 36; ++it) {
    int c = 2*it + g;
    kindL[nl*KSTR + c] = (unsigned char)dots[c*NBOARD + boardBase + nl];
  }
  __syncthreads();

  auto loadB = [&](int off, int NT, int kc, int nt) -> v16h {
    return *(const v16h*)(blob + off + (((kc*NT + nt) << 5) + lane) * 16);
  };
  auto maskA = [&](int kind, int kc) -> v16h {        // A = (d == kind)
    v16h a;
    #pragma unroll
    for (int t = 0; t < 16; ++t) {
      int c = kc*32 + kidxA(t, g);
      float v = 0.0f;
      if (c < 72) v = (kindL[nl*KSTR + c] == kind) ? 1.0f : 0.0f;
      a[t] = (_Float16)v;
    }
    return a;
  };
  auto notA = [&](int kind, int kc) -> v16h {         // A = (mask+empty > 0) ? 0 : 1
    v16h a;
    #pragma unroll
    for (int t = 0; t < 16; ++t) {
      int c = kc*32 + kidxA(t, g);
      float v = 0.0f;
      if (c < 72) {
        float s = ((kindL[nl*KSTR + c] == kind) ? 1.0f : 0.0f) + EL[nl*FSTR + c];
        v = (s > 0.0f) ? 0.0f : 1.0f;
      }
      a[t] = (_Float16)v;
    }
    return a;
  };
  auto not2A = [&](int kind, int kc) -> v16h {        // A = (mask+empty > 0) ? 0 : acv2
    v16h a;
    #pragma unroll
    for (int t = 0; t < 16; ++t) {
      int c = kc*32 + kidxA(t, g);
      float v = 0.0f;
      if (c < 72) {
        float s = ((kindL[nl*KSTR + c] == kind) ? 1.0f : 0.0f) + EL[nl*FSTR + c];
        v = (s > 0.0f) ? 0.0f : AL[nl*FSTR + c];
      }
      a[t] = (_Float16)v;
    }
    return a;
  };
  auto gemm3 = [&](v16h a0, v16h a1, v16h a2, int off, int nt, float bias) -> v8f {
    v8f acc;
    #pragma unroll
    for (int r = 0; r < 8; ++r) acc[r] = bias;
    acc = wmma16(a0, loadB(off, 5, 0, nt), acc);
    acc = wmma16(a1, loadB(off, 5, 1, nt), acc);
    acc = wmma16(a2, loadB(off, 5, 2, nt), acc);
    return acc;
  };

  const float bEach  = *b_each_p;
  const float bNot   = *b_not_p;
  const float bNot2  = *b_not2_p;
  const float bEmpty = *b_empty_p;

  // ---- empty = Mempty @ (d==0) + b  -> LDS (C/D layout transpose) ----
  {
    v16h e0 = maskA(0, 0), e1 = maskA(0, 1), e2 = maskA(0, 2);
    #pragma unroll
    for (int nt = 0; nt < 5; ++nt) {
      v8f acc = gemm3(e0, e1, e2, OFF_EMPTY, nt, bEmpty);
      #pragma unroll
      for (int r = 0; r < 8; ++r) EL[(r + 8*g)*FSTR + nt*16 + nl] = acc[r];
    }
  }

  // ---- stage 1: all_color_value = sum_k mask_k * (ecv_k + empty - notconv_k) ----
  v8f acv[5];
  #pragma unroll
  for (int nt = 0; nt < 5; ++nt)
    #pragma unroll
    for (int r = 0; r < 8; ++r) acv[nt][r] = 0.0f;

  for (int k = 1; k <= 4; ++k) {
    v16h m0 = maskA(k, 0), m1 = maskA(k, 1), m2 = maskA(k, 2);
    v16h q0 = notA(k, 0),  q1 = notA(k, 1),  q2 = notA(k, 2);
    #pragma unroll
    for (int nt = 0; nt < 5; ++nt) {
      v8f ec = gemm3(m0, m1, m2, OFF_EACH, nt, bEach);
      v8f nc = gemm3(q0, q1, q2, OFF_NOT,  nt, bNot);
      #pragma unroll
      for (int r = 0; r < 8; ++r) {
        int m = r + 8*g, n = nt*16 + nl;
        float val = ec[r] + EL[m*FSTR + n] - nc[r];
        bool on = (n < 72) && (kindL[m*KSTR + n] == k);
        acv[nt][r] += on ? val : 0.0f;
      }
    }
  }
  #pragma unroll
  for (int nt = 0; nt < 5; ++nt)
    #pragma unroll
    for (int r = 0; r < 8; ++r)
      AL[(r + 8*g)*FSTR + nt*16 + nl] = acv[nt][r];

  // ---- stage 2: sequential scan over kinds (carry acv2 in LDS; A-frags read before updates) ----
  for (int k = 1; k <= 4; ++k) {
    v16h m0 = maskA(k, 0), m1 = maskA(k, 1), m2 = maskA(k, 2);
    v16h q0 = not2A(k, 0), q1 = not2A(k, 1), q2 = not2A(k, 2);
    #pragma unroll
    for (int nt = 0; nt < 5; ++nt) {
      v8f ec = gemm3(m0, m1, m2, OFF_EACH, nt, bEach);
      v8f c2 = gemm3(q0, q1, q2, OFF_NOT2, nt, bNot2);
      #pragma unroll
      for (int r = 0; r < 8; ++r) {
        int m = r + 8*g, n = nt*16 + nl;
        float v = ec[r] + EL[m*FSTR + n] + c2[r];
        bool on = (n < 72) && (kindL[m*KSTR + n] == k);
        AL[m*FSTR + n] += on ? v : 0.0f;
      }
    }
  }

  // ---- MLP: 72 -> 100 -> 100 -> 1, leaky_relu(0.2) ----
  auto actA = [&](const float* src, int stride, int kmax, int kc) -> v16h {
    v16h a;
    #pragma unroll
    for (int t = 0; t < 16; ++t) {
      int c = kc*32 + kidxA(t, g);
      float v = (c < kmax) ? src[nl*stride + c] : 0.0f;
      a[t] = (_Float16)v;
    }
    return a;
  };
  { // layer 1
    v16h x0 = actA(AL, FSTR, 72, 0), x1 = actA(AL, FSTR, 72, 1), x2 = actA(AL, FSTR, 72, 2);
    #pragma unroll
    for (int nt = 0; nt < 7; ++nt) {
      v8f acc;
      #pragma unroll
      for (int r = 0; r < 8; ++r) acc[r] = biasL[nt*16 + nl];
      acc = wmma16(x0, loadB(OFF_W1, 7, 0, nt), acc);
      acc = wmma16(x1, loadB(OFF_W1, 7, 1, nt), acc);
      acc = wmma16(x2, loadB(OFF_W1, 7, 2, nt), acc);
      #pragma unroll
      for (int r = 0; r < 8; ++r) {
        float v = acc[r]; v = (v > 0.0f) ? v : 0.2f*v;
        HL[(r + 8*g)*HSTR + nt*16 + nl] = v;
      }
    }
  }
  { // layer 2 (all A-frag reads precede HL overwrite; per-wave DS is in-order)
    v16h h0 = actA(HL, HSTR, 100, 0), h1 = actA(HL, HSTR, 100, 1),
         h2 = actA(HL, HSTR, 100, 2), h3 = actA(HL, HSTR, 100, 3);
    #pragma unroll
    for (int nt = 0; nt < 7; ++nt) {
      v8f acc;
      #pragma unroll
      for (int r = 0; r < 8; ++r) acc[r] = biasL[112 + nt*16 + nl];
      acc = wmma16(h0, loadB(OFF_W2, 7, 0, nt), acc);
      acc = wmma16(h1, loadB(OFF_W2, 7, 1, nt), acc);
      acc = wmma16(h2, loadB(OFF_W2, 7, 2, nt), acc);
      acc = wmma16(h3, loadB(OFF_W2, 7, 3, nt), acc);
      #pragma unroll
      for (int r = 0; r < 8; ++r) {
        float v = acc[r]; v = (v > 0.0f) ? v : 0.2f*v;
        HL[(r + 8*g)*HSTR + nt*16 + nl] = v;
      }
    }
  }
  { // layer 3 -> one scalar per board (column n==0 of the C/D tile)
    v16h z0 = actA(HL, HSTR, 100, 0), z1 = actA(HL, HSTR, 100, 1),
         z2 = actA(HL, HSTR, 100, 2), z3 = actA(HL, HSTR, 100, 3);
    v8f acc;
    #pragma unroll
    for (int r = 0; r < 8; ++r) acc[r] = biasL[224];
    acc = wmma16(z0, loadB(OFF_W3, 1, 0, 0), acc);
    acc = wmma16(z1, loadB(OFF_W3, 1, 1, 0), acc);
    acc = wmma16(z2, loadB(OFF_W3, 1, 2, 0), acc);
    acc = wmma16(z3, loadB(OFF_W3, 1, 3, 0), acc);
    if (nl == 0) {
      #pragma unroll
      for (int r = 0; r < 8; ++r) out[boardBase + r + 8*g] = acc[r];
    }
  }
}

extern "C" void kernel_launch(void* const* d_in, const int* in_sizes, int n_in,
                              void* d_out, int out_size, void* d_ws, size_t ws_size,
                              hipStream_t stream) {
  const int*   dots    = (const int*)  d_in[0];
  const float* w_each  = (const float*)d_in[1];
  const float* b_each  = (const float*)d_in[2];
  const float* w_not   = (const float*)d_in[3];
  const float* b_not   = (const float*)d_in[4];
  const float* w_not2  = (const float*)d_in[5];
  const float* b_not2  = (const float*)d_in[6];
  const float* w_empty = (const float*)d_in[7];
  const float* b_empty = (const float*)d_in[8];
  const float* W1 = (const float*)d_in[9];
  const float* b1 = (const float*)d_in[10];
  const float* W2 = (const float*)d_in[11];
  const float* b2 = (const float*)d_in[12];
  const float* W3 = (const float*)d_in[13];
  const float* b3 = (const float*)d_in[14];

  (void)hipFuncSetAttribute((const void*)board_mlp_kernel,
                            hipFuncAttributeMaxDynamicSharedMemorySize, LDS_TOTAL);

  const int prepElems = BLOB_HW + BIAS_FLOATS;
  prep_kernel<<<(prepElems + 255)/256, 256, 0, stream>>>(
      w_each, w_not, w_not2, w_empty, W1, W2, W3, b1, b2, b3, d_ws);

  const int blocks = NBOARD / (WAVES * 16);   // 512
  board_mlp_kernel<<<blocks, TPB, LDS_TOTAL, stream>>>(
      dots, b_each, b_not, b_not2, b_empty, d_ws, (float*)d_out);
}
